// DynamicFusedMOE_4861902979824
// MI455X (gfx1250) — compile-verified, matrix operand order checked by reading, not run on hardware
//
#include <hip/hip_runtime.h>
#include <hip/hip_bf16.h>
#include <stdint.h>

// ---------------- problem constants ----------------
#define NE   8
#define DM   2048
#define DFF  4096
#define TT   2048
#define TOPK 2
#define PP   (TT * TOPK)   // 4096 routed rows

// ---------------- tile constants ----------------
#define BM   128
#define BN   128
#define BK   32
#define LDT  40            // LDS row stride in bf16 elems: 80B, 16B-aligned, conflict-free

typedef __attribute__((ext_vector_type(16))) __bf16 v16bf;
typedef __attribute__((ext_vector_type(8)))  __bf16 v8bf;
typedef __attribute__((ext_vector_type(8)))  float  v8f;

__device__ __forceinline__ int imin(int a, int b) { return a < b ? a : b; }

__device__ __forceinline__ v16bf ldfrag(const __bf16* p) {
  union { v16bf v; v8bf h[2]; } f;
  f.h[0] = *(const v8bf*)(p);
  f.h[1] = *(const v8bf*)(p + 16);
  return f.v;
}

__device__ __forceinline__ v8bf pack8(float4 a, float4 b) {
  v8bf r;
  r[0] = (__bf16)a.x; r[1] = (__bf16)a.y; r[2] = (__bf16)a.z; r[3] = (__bf16)a.w;
  r[4] = (__bf16)b.x; r[5] = (__bf16)b.y; r[6] = (__bf16)b.z; r[7] = (__bf16)b.w;
  return r;
}

// convert 16 fp32 -> 16 bf16 into LDS at dst (16B aligned)
__device__ __forceinline__ void cvt_store16(__bf16* dst, const float* src) {
  const float4* s = (const float4*)src;
  float4 a = s[0], b = s[1], c = s[2], d = s[3];
  *(v8bf*)(dst)     = pack8(a, b);
  *(v8bf*)(dst + 8) = pack8(c, d);
}

#define WMMA_BF16(A, B, C) \
  __builtin_amdgcn_wmma_f32_16x16x32_bf16(false, (A), false, (B), (short)0, (C), false, false)

// -------- gfx1250 async global->LDS path (bf16 tiles, no conversion needed) --------
#if defined(__gfx1250__) && __has_builtin(__builtin_amdgcn_global_load_async_to_lds_b128)
#define USE_ASYNC 1
typedef int v4i __attribute__((vector_size(4 * sizeof(int))));   // matches builtin's param type
typedef __attribute__((address_space(1))) v4i* as1_v4i_p;        // global  (prints as __device__)
typedef __attribute__((address_space(3))) v4i* as3_v4i_p;        // shared  (LDS)
__device__ __forceinline__ void async_b128(const void* g, void* l) {
  // flat->addrspace via integer cast: LDS flat addr low 32 bits == LDS offset (ISA aperture rule)
  __builtin_amdgcn_global_load_async_to_lds_b128(
      (as1_v4i_p)(uintptr_t)g, (as3_v4i_p)(uint32_t)(uintptr_t)l, 0, 0);
}
__device__ __forceinline__ void async_wait0() { __builtin_amdgcn_s_wait_asynccnt(0); }
#else
#define USE_ASYNC 0
#endif

// ==================== routing ====================
__global__ void k_route(const float* __restrict__ score,
                        int* __restrict__ top_e, float* __restrict__ top_c) {
  int t = blockIdx.x * blockDim.x + threadIdx.x;
  if (t >= TT) return;
  float s[NE];
#pragma unroll
  for (int i = 0; i < NE; ++i) s[i] = score[t * NE + i];
  int e0 = 0; float v0 = s[0];
#pragma unroll
  for (int i = 1; i < NE; ++i) if (s[i] > v0) { v0 = s[i]; e0 = i; }
  int e1 = -1; float v1 = -3.4e38f;
#pragma unroll
  for (int i = 0; i < NE; ++i) if (i != e0 && s[i] > v1) { v1 = s[i]; e1 = i; }
  float x1 = __expf(v1 - v0);           // x0 == 1; softmax denom cancels on renorm
  float inv = __builtin_amdgcn_rcpf(1.0f + x1);
  top_e[2 * t + 0] = e0;  top_e[2 * t + 1] = e1;
  top_c[2 * t + 0] = inv; top_c[2 * t + 1] = x1 * inv;
}

__global__ void k_count(const int* __restrict__ top_e, int* __restrict__ cnt) {
  int e = blockIdx.x;
  int lane = threadIdx.x;   // one wave32
  int c = 0;
  for (int t = lane; t < TT; t += 32)
    c += (top_e[2 * t] == e) + (top_e[2 * t + 1] == e);
#pragma unroll
  for (int o = 16; o > 0; o >>= 1) c += __shfl_down(c, o, 32);
  if (lane == 0) cnt[e] = c;
}

__global__ void k_scan(const int* __restrict__ cnt, int* __restrict__ off) {
  if (threadIdx.x == 0) {
    int acc = 0;
#pragma unroll
    for (int e = 0; e < NE; ++e) { off[e] = acc; acc += cnt[e]; }
    off[NE] = acc;
  }
}

__global__ void k_fill(const int* __restrict__ top_e, const int* __restrict__ off,
                       int* __restrict__ row_token, int* __restrict__ token_rows) {
  int e = blockIdx.x;
  int lane = threadIdx.x;   // one wave32
  int run = off[e];
  unsigned ltmask = (1u << lane) - 1u;
  for (int t0 = 0; t0 < TT; t0 += 32) {
    int t = t0 + lane;
#pragma unroll
    for (int s = 0; s < 2; ++s) {
      bool match = (top_e[2 * t + s] == e);
      unsigned mb = (unsigned)__ballot(match);
      if (match) {
        int idx = run + __popc(mb & ltmask);
        row_token[idx] = t;
        token_rows[2 * t + s] = idx;
      }
      run += __popc(mb);
    }
  }
}

// ==================== stage-1 grouped GEMM: h = silu(X W1g^T) * (X W1u^T) ====================
// grid: (DFF/BN, TT/BM, NE), block: 256.  Double-buffered LDS, 1 barrier / K-step.
__global__ __launch_bounds__(256)
void k_gemm1(const float* __restrict__ x, const float* __restrict__ w1,
             const int* __restrict__ row_token, const int* __restrict__ off,
             const int* __restrict__ cnt_, __bf16* __restrict__ h) {
  const int e = blockIdx.z;
  const int cnt = cnt_[e];
  const int m0 = blockIdx.y * BM;
  if (m0 >= cnt) return;
  const int n0 = blockIdx.x * BN;
  const int base = off[e];
  const float* w1e = w1 + (size_t)e * (2 * DFF) * DM;

  __shared__ __align__(16) __bf16 As[2][BM * LDT];
  __shared__ __align__(16) __bf16 Bg[2][BN * LDT];
  __shared__ __align__(16) __bf16 Bu[2][BN * LDT];

  const int tid  = threadIdx.x;
  const int lrow = tid >> 1;          // 0..127
  const int seg  = (tid & 1) * 16;    // 0 | 16 (elements)

  const int gr  = m0 + lrow;
  const int tok = row_token[base + imin(gr, cnt - 1)];
  const float* aSrc = x   + (size_t)tok * DM + seg;
  const float* gSrc = w1e + (size_t)(n0 + lrow) * DM + seg;
  const float* uSrc = w1e + (size_t)(DFF + n0 + lrow) * DM + seg;

  const int lane = tid & 31;
  const int wave = tid >> 5;
  const int wm = (wave & 3) * 32;     // 4 waves in M
  const int wn = (wave >> 2) * 64;    // 2 waves in N
  const int r  = lane & 15;
  const int kb = (lane >> 4) * 8;

  v8f accG[2][4], accU[2][4];
#pragma unroll
  for (int mi = 0; mi < 2; ++mi)
#pragma unroll
    for (int ni = 0; ni < 4; ++ni) { accG[mi][ni] = {}; accU[mi][ni] = {}; }

  const int NIT = DM / BK;            // 64
  float aBuf[16], gBuf[16], uBuf[16];

  // prologue: tile 0 -> buf 0
#pragma unroll
  for (int i = 0; i < 4; ++i) {
    *(float4*)(aBuf + 4 * i) = *(const float4*)(aSrc + 4 * i);
    *(float4*)(gBuf + 4 * i) = *(const float4*)(gSrc + 4 * i);
    *(float4*)(uBuf + 4 * i) = *(const float4*)(uSrc + 4 * i);
  }
  cvt_store16(&As[0][lrow * LDT + seg], aBuf);
  cvt_store16(&Bg[0][lrow * LDT + seg], gBuf);
  cvt_store16(&Bu[0][lrow * LDT + seg], uBuf);
  __syncthreads();

  for (int it = 0; it < NIT; ++it) {
    const int cur = it & 1;
    const int nxt = cur ^ 1;
    const bool more = (it + 1) < NIT;
    if (more) {
      const int k0 = (it + 1) * BK;
#pragma unroll
      for (int i = 0; i < 4; ++i) {
        *(float4*)(aBuf + 4 * i) = *(const float4*)(aSrc + k0 + 4 * i);
        *(float4*)(gBuf + 4 * i) = *(const float4*)(gSrc + k0 + 4 * i);
        *(float4*)(uBuf + 4 * i) = *(const float4*)(uSrc + k0 + 4 * i);
      }
    }

    v16bf a0 = ldfrag(&As[cur][(wm +  0 + r) * LDT + kb]);
    v16bf a1 = ldfrag(&As[cur][(wm + 16 + r) * LDT + kb]);
#pragma unroll
    for (int ni = 0; ni < 4; ++ni) {
      v16bf bg = ldfrag(&Bg[cur][(wn + ni * 16 + r) * LDT + kb]);
      v16bf bu = ldfrag(&Bu[cur][(wn + ni * 16 + r) * LDT + kb]);
      accG[0][ni] = WMMA_BF16(a0, bg, accG[0][ni]);
      accG[1][ni] = WMMA_BF16(a1, bg, accG[1][ni]);
      accU[0][ni] = WMMA_BF16(a0, bu, accU[0][ni]);
      accU[1][ni] = WMMA_BF16(a1, bu, accU[1][ni]);
    }

    if (more) {
      cvt_store16(&As[nxt][lrow * LDT + seg], aBuf);
      cvt_store16(&Bg[nxt][lrow * LDT + seg], gBuf);
      cvt_store16(&Bu[nxt][lrow * LDT + seg], uBuf);
    }
    __syncthreads();
  }

  // epilogue: h = silu(g) * u, bf16
  const int nlane = lane & 15;
  const int mhalf = (lane >> 4) * 8;
#pragma unroll
  for (int mi = 0; mi < 2; ++mi)
#pragma unroll
    for (int ni = 0; ni < 4; ++ni) {
      int nCol = n0 + wn + ni * 16 + nlane;
#pragma unroll
      for (int v = 0; v < 8; ++v) {
        int mRow = wm + mi * 16 + mhalf + v;
        float g = accG[mi][ni][v], u = accU[mi][ni][v];
        float sv = g * __builtin_amdgcn_rcpf(1.0f + __expf(-g)) * u;
        if (m0 + mRow < cnt)
          h[(size_t)(base + m0 + mRow) * DFF + nCol] = (__bf16)sv;
      }
    }
}

// ==================== stage-2 grouped GEMM: yslot = h W2^T ====================
// grid: (DM/BN, TT/BM, NE), block: 256.  A tile (bf16) uses async global->LDS when available.
__global__ __launch_bounds__(256)
void k_gemm2(const __bf16* __restrict__ h, const float* __restrict__ w2,
             const int* __restrict__ off, const int* __restrict__ cnt_,
             float* __restrict__ yslot) {
  const int e = blockIdx.z;
  const int cnt = cnt_[e];
  const int m0 = blockIdx.y * BM;
  if (m0 >= cnt) return;
  const int n0 = blockIdx.x * BN;
  const int base = off[e];
  const float* w2e = w2 + (size_t)e * DM * DFF;

  __shared__ __align__(16) __bf16 As[2][BM * LDT];
  __shared__ __align__(16) __bf16 Bs[2][BN * LDT];

  const int tid  = threadIdx.x;
  const int lrow = tid >> 1;
  const int seg  = (tid & 1) * 16;

  const int arow = base + imin(m0 + lrow, cnt - 1);
  const __bf16* aSrc = h   + (size_t)arow * DFF + seg;
  const float*  bSrc = w2e + (size_t)(n0 + lrow) * DFF + seg;

  const int lane = tid & 31;
  const int wave = tid >> 5;
  const int wm = (wave & 3) * 32;
  const int wn = (wave >> 2) * 64;
  const int r  = lane & 15;
  const int kb = (lane >> 4) * 8;

  v8f acc[2][4];
#pragma unroll
  for (int mi = 0; mi < 2; ++mi)
#pragma unroll
    for (int ni = 0; ni < 4; ++ni) acc[mi][ni] = {};

  const int NIT = DFF / BK;           // 128
  float bBuf[16];
#if !USE_ASYNC
  v8bf aReg0, aReg1;
#endif

  // prologue: tile 0 -> buf 0
#if USE_ASYNC
  async_b128(aSrc,     &As[0][lrow * LDT + seg]);
  async_b128(aSrc + 8, &As[0][lrow * LDT + seg + 8]);
#else
  aReg0 = *(const v8bf*)(aSrc);
  aReg1 = *(const v8bf*)(aSrc + 8);
  *(v8bf*)(&As[0][lrow * LDT + seg])     = aReg0;
  *(v8bf*)(&As[0][lrow * LDT + seg + 8]) = aReg1;
#endif
#pragma unroll
  for (int i = 0; i < 4; ++i)
    *(float4*)(bBuf + 4 * i) = *(const float4*)(bSrc + 4 * i);
  cvt_store16(&Bs[0][lrow * LDT + seg], bBuf);
#if USE_ASYNC
  async_wait0();
#endif
  __syncthreads();

  for (int it = 0; it < NIT; ++it) {
    const int cur = it & 1;
    const int nxt = cur ^ 1;
    const bool more = (it + 1) < NIT;
    if (more) {
      const int k0 = (it + 1) * BK;
#if USE_ASYNC
      async_b128(aSrc + k0,     &As[nxt][lrow * LDT + seg]);
      async_b128(aSrc + k0 + 8, &As[nxt][lrow * LDT + seg + 8]);
#else
      aReg0 = *(const v8bf*)(aSrc + k0);
      aReg1 = *(const v8bf*)(aSrc + k0 + 8);
#endif
#pragma unroll
      for (int i = 0; i < 4; ++i)
        *(float4*)(bBuf + 4 * i) = *(const float4*)(bSrc + k0 + 4 * i);
    }

    v16bf fa0 = ldfrag(&As[cur][(wm +  0 + r) * LDT + kb]);
    v16bf fa1 = ldfrag(&As[cur][(wm + 16 + r) * LDT + kb]);
#pragma unroll
    for (int ni = 0; ni < 4; ++ni) {
      v16bf fb = ldfrag(&Bs[cur][(wn + ni * 16 + r) * LDT + kb]);
      acc[0][ni] = WMMA_BF16(fa0, fb, acc[0][ni]);
      acc[1][ni] = WMMA_BF16(fa1, fb, acc[1][ni]);
    }

    if (more) {
#if !USE_ASYNC
      *(v8bf*)(&As[nxt][lrow * LDT + seg])     = aReg0;
      *(v8bf*)(&As[nxt][lrow * LDT + seg + 8]) = aReg1;
#endif
      cvt_store16(&Bs[nxt][lrow * LDT + seg], bBuf);
    }
#if USE_ASYNC
    async_wait0();
#endif
    __syncthreads();
  }

  const int nlane = lane & 15;
  const int mhalf = (lane >> 4) * 8;
#pragma unroll
  for (int mi = 0; mi < 2; ++mi)
#pragma unroll
    for (int ni = 0; ni < 4; ++ni) {
      int nCol = n0 + wn + ni * 16 + nlane;
#pragma unroll
      for (int v = 0; v < 8; ++v) {
        int mRow = wm + mi * 16 + mhalf + v;
        if (m0 + mRow < cnt)
          yslot[(size_t)(base + m0 + mRow) * DM + nCol] = acc[mi][ni][v];
      }
    }
}

// ==================== combine: out[t] = c0*y[r0] + c1*y[r1] ====================
__global__ void k_combine(const float* __restrict__ yslot,
                          const int* __restrict__ token_rows,
                          const float* __restrict__ top_c,
                          float* __restrict__ out) {
  int t  = blockIdx.y;
  int c4 = blockIdx.x * blockDim.x + threadIdx.x;
  int r0 = token_rows[2 * t], r1 = token_rows[2 * t + 1];
  float c0 = top_c[2 * t], c1 = top_c[2 * t + 1];
  const float4* y0 = (const float4*)(yslot + (size_t)r0 * DM);
  const float4* y1 = (const float4*)(yslot + (size_t)r1 * DM);
  float4 a = y0[c4], b = y1[c4];
  float4 o;
  o.x = c0 * a.x + c1 * b.x; o.y = c0 * a.y + c1 * b.y;
  o.z = c0 * a.z + c1 * b.z; o.w = c0 * a.w + c1 * b.w;
  ((float4*)(out + (size_t)t * DM))[c4] = o;
}

// ==================== launch ====================
extern "C" void kernel_launch(void* const* d_in, const int* in_sizes, int n_in,
                              void* d_out, int out_size, void* d_ws, size_t ws_size,
                              hipStream_t stream) {
  const float* x     = (const float*)d_in[0];
  const float* w1    = (const float*)d_in[1];
  const float* w2    = (const float*)d_in[2];
  const float* score = (const float*)d_in[3];
  float* out = (float*)d_out;

  char* ws = (char*)d_ws;
  __bf16* h      = (__bf16*)ws;                 ws += (size_t)PP * DFF * sizeof(__bf16); // 32MB
  float*  yslot  = (float*)ws;                  ws += (size_t)PP * DM * sizeof(float);   // 32MB
  int*    top_e  = (int*)ws;                    ws += (size_t)TT * 2 * sizeof(int);
  float*  top_c  = (float*)ws;                  ws += (size_t)TT * 2 * sizeof(float);
  int*    row_token  = (int*)ws;                ws += (size_t)PP * sizeof(int);
  int*    token_rows = (int*)ws;                ws += (size_t)TT * 2 * sizeof(int);
  int*    cnt    = (int*)ws;                    ws += (size_t)(NE) * sizeof(int);
  int*    off    = (int*)ws;                    ws += (size_t)(NE + 1) * sizeof(int);

  k_route<<<dim3((TT + 255) / 256), dim3(256), 0, stream>>>(score, top_e, top_c);
  k_count<<<dim3(NE), dim3(32), 0, stream>>>(top_e, cnt);
  k_scan <<<dim3(1), dim3(1), 0, stream>>>(cnt, off);
  k_fill <<<dim3(NE), dim3(32), 0, stream>>>(top_e, off, row_token, token_rows);

  k_gemm1<<<dim3(DFF / BN, TT / BM, NE), dim3(256), 0, stream>>>(x, w1, row_token, off, cnt, h);
  k_gemm2<<<dim3(DM / BN, TT / BM, NE), dim3(256), 0, stream>>>(h, w2, off, cnt, yslot);
  k_combine<<<dim3(DM / 4 / 256, TT), dim3(256), 0, stream>>>(yslot, token_rows, top_c, out);
}
